// W4A16Linear_68951404970078
// MI455X (gfx1250) — matrix operand order repaired and checked by reading, not compiled, not run on hardware
//
#include <hip/hip_runtime.h>

// ---------------------------------------------------------------------------
// W4A16 (NVFP4 weight, bf16 activation) GEMM for gfx1250 using WMMA bf16.
//   out[M,N] = (bf16)x[M,K]  x  dequant(wq)[N,K]^T,  fp32 output
//   M=2048, N=4096, K=4096; block-scale every 16 elements along K.
// 256x128 tile per 256-thread (8 wave32) block, 64x64 wave tile (16 WMMA/wave),
// K-step 32, double-buffered LDS, branchless nibble-LUT NVFP4 decode.
// ---------------------------------------------------------------------------

typedef __bf16 bf16;
typedef __attribute__((ext_vector_type(16))) __bf16 v16bf;
typedef __attribute__((ext_vector_type(8)))  float  v8f;
typedef unsigned int u32x4 __attribute__((ext_vector_type(4)));
typedef float        f32x4 __attribute__((ext_vector_type(4)));
typedef int          i32x4 __attribute__((ext_vector_type(4)));

constexpr int Mdim = 2048;
constexpr int Ndim = 4096;
constexpr int Kdim = 4096;

constexpr int TM   = 256;   // block tile M (A rows)
constexpr int TN   = 128;   // block tile N (W rows)
constexpr int TK   = 32;    // K step (matches 16x16x32 bf16 WMMA)
constexpr int LSTR = 40;    // LDS row stride in bf16 (32 + 8 pad -> 80B, 16B aligned)
constexpr int NTHREADS = 256; // 8 wave32

// Branchless NVFP4 (e2m1) decode * (0.5*scale).
// Magnitudes*2 for mag-codes 0..7 are {0,1,2,3,4,6,8,12} -> nibbles 0xC8643210.
// Sign folded by OR into the float sign bit (value >= 0 since scale > 0).
__device__ __forceinline__ float nvfp4_dec_hs(int c, float halfScale) {
    unsigned mag2 = (0xC8643210u >> ((c << 2) & 0x1C)) & 0xFu;  // 2*|value|
    float v = (float)mag2 * halfScale;                          // |value| * scale
    return __uint_as_float(__float_as_uint(v) | (((unsigned)c & 8u) << 28));
}

union P16 { bf16 h[16]; u32x4 q[2]; };       // 32B staging pack
union Frag { v16bf v; u32x4 q[2]; };         // WMMA A/B fragment (8 VGPRs)

__global__ __launch_bounds__(NTHREADS)
void w4a16_wmma_kernel(const float* __restrict__ x,
                       const float* __restrict__ wscale,
                       const int*   __restrict__ wq,
                       float* __restrict__ out)
{
    __shared__ __align__(16) bf16 As[2][TM * LSTR];   // x tile, bf16 [row=m][k]
    __shared__ __align__(16) bf16 Bs[2][TN * LSTR];   // w tile, bf16 [row=o][k]

    const int tid  = threadIdx.x;
    const int lane = tid & 31;
    const int wv   = tid >> 5;            // 0..7

    const int mBase = blockIdx.y * TM;
    const int nBase = blockIdx.x * TN;

    // ---- A staging: 1 thread per row (256 rows), 32 fp32 each ----
    const int aRow = tid;                 // 0..255
    const float* xP = x + (size_t)(mBase + aRow) * Kdim;

    // ---- B staging: 2 threads per row (128 rows), 16 codes each ----
    const int bRow = tid >> 1;            // 0..127
    const int bK   = (tid & 1) << 4;      // 0 or 16 (each within one scale block)
    const int*   qP = wq + (size_t)(nBase + bRow) * Kdim + bK;
    const float* sP = wscale + (size_t)(nBase + bRow) * (Kdim / 16) + (bK >> 4);

    f32x4 aR[8];      // 32 fp32 activations in flight
    i32x4 qR[4];      // 16 weight codes in flight
    float sR;         // block scale for this thread's 16 codes

    auto loadGlobal = [&](int kt) {
        const f32x4* ap = (const f32x4*)(xP + kt * TK);
#pragma unroll
        for (int i = 0; i < 8; ++i) aR[i] = ap[i];
        const i32x4* qp = (const i32x4*)(qP + kt * TK);
#pragma unroll
        for (int i = 0; i < 4; ++i) qR[i] = qp[i];
        sR = sP[kt * (TK / 16)];
    };

    auto storeLds = [&](int buf) {
        // A: fp32 -> bf16 (lowered to v_cvt_pk_bf16_f32 pairs), 32 elems
        const float* af = (const float*)aR;
        bf16* ad = &As[buf][aRow * LSTR];
#pragma unroll
        for (int h = 0; h < 2; ++h) {
            P16 pa;
#pragma unroll
            for (int i = 0; i < 16; ++i) pa.h[i] = (bf16)af[h * 16 + i];
            *(u32x4*)(ad + h * 16)     = pa.q[0];
            *(u32x4*)(ad + h * 16 + 8) = pa.q[1];
        }
        // B: branchless fp4 decode * scale -> bf16, 16 elems
        P16 pb;
        const int* qi = (const int*)qR;
        const float hs = 0.5f * sR;
#pragma unroll
        for (int i = 0; i < 16; ++i) pb.h[i] = (bf16)nvfp4_dec_hs(qi[i], hs);
        bf16* bd = &Bs[buf][bRow * LSTR + bK];
        *(u32x4*)(bd)     = pb.q[0];
        *(u32x4*)(bd + 8) = pb.q[1];
    };

    // ---- per-wave compute layout: wave tile 64(M) x 64(N); 4 x 2 wave grid ----
    const int wm = (wv & 3) * 64;         // 0,64,128,192
    const int wn = (wv >> 2) * 64;        // 0,64
    const int rA = lane & 15;             // row/col within 16-wide tile
    const int hi = (lane >> 4) & 1;       // lane half
    const int kA = hi * 8;                // A frag: K {0..7,16..23} vs {8..15,24..31}
    const int kB = hi * 8;                // B frag: K {0..7,16..23} vs {8..15,24..31}

    v8f acc[4][4] = {};                   // 16 accumulators = 128 f32 VGPRs

    loadGlobal(0);
    storeLds(0);

    constexpr int nK = Kdim / TK;         // 128 K-steps
    for (int kt = 0; kt < nK; ++kt) {
        __syncthreads();
        const int cur = kt & 1;

        if (kt + 1 < nK) loadGlobal(kt + 1);   // overlap next tile's global loads

        Frag a[4], b[4];
#pragma unroll
        for (int mi = 0; mi < 4; ++mi) {
            const bf16* p = &As[cur][(wm + mi * 16 + rA) * LSTR];
            a[mi].q[0] = *(const u32x4*)(p + kA);
            a[mi].q[1] = *(const u32x4*)(p + 16 + kA);
        }
#pragma unroll
        for (int ni = 0; ni < 4; ++ni) {
            const bf16* p = &Bs[cur][(wn + ni * 16 + rA) * LSTR];
            b[ni].q[0] = *(const u32x4*)(p + kB);
            b[ni].q[1] = *(const u32x4*)(p + 16 + kB);
        }

#pragma unroll
        for (int mi = 0; mi < 4; ++mi)
#pragma unroll
            for (int ni = 0; ni < 4; ++ni)
                acc[mi][ni] = __builtin_amdgcn_wmma_f32_16x16x32_bf16(
                    /*neg_a=*/false, a[mi].v, /*neg_b=*/false, b[ni].v,
                    /*c_mod=*/(short)0, acc[mi][ni],
                    /*reuse_a=*/false, /*reuse_b=*/false);

        if (kt + 1 < nK) storeLds((kt + 1) & 1);  // fill the other buffer
    }

    // ---- write out (round through bf16 to match reference, store fp32) ----
    const int rowHalf = hi * 8;
#pragma unroll
    for (int mi = 0; mi < 4; ++mi) {
#pragma unroll
        for (int ni = 0; ni < 4; ++ni) {
            const int col = nBase + wn + ni * 16 + rA;
#pragma unroll
            for (int r = 0; r < 8; ++r) {
                const int row = mBase + wm + mi * 16 + rowHalf + r;
                out[(size_t)row * Ndim + col] = (float)(bf16)acc[mi][ni][r];
            }
        }
    }
}

extern "C" void kernel_launch(void* const* d_in, const int* in_sizes, int n_in,
                              void* d_out, int out_size, void* d_ws, size_t ws_size,
                              hipStream_t stream) {
    (void)in_sizes; (void)n_in; (void)out_size; (void)d_ws; (void)ws_size;
    const float* x      = (const float*)d_in[0];
    const float* wscale = (const float*)d_in[1];
    const int*   wq     = (const int*)d_in[2];
    float*       out    = (float*)d_out;

    dim3 grid(Ndim / TN, Mdim / TM);   // (32, 8) = 256 workgroups
    dim3 block(NTHREADS);              // 8 wave32 per workgroup
    w4a16_wmma_kernel<<<grid, block, 0, stream>>>(x, wscale, wq, out);
}
